// DeepSC_10136122819141
// MI455X (gfx1250) — compile-verified, matrix operand order checked by reading, not run on hardware
//
#include <hip/hip_runtime.h>
#include <hip/hip_bf16.h>

#define DEV __device__ __forceinline__

typedef __bf16 bf16;
typedef __attribute__((ext_vector_type(16))) __bf16 v16bf;
typedef __attribute__((ext_vector_type(8)))  __bf16 v8bf;
typedef __attribute__((ext_vector_type(8)))  float  v8f;

constexpr int kDim         = 1024;    // model dim
constexpr int kInter       = 1024;    // routed expert inter dim
constexpr int kE           = 8;       // routed experts
constexpr int kSharedInter = 2048;    // shared expert inter dim
constexpr int kT           = 4 * 2048;// tokens = B*S

// LDS geometry: rows padded to 80 bytes (20 dwords -> conflict-free b128 reads).
// One pipeline stage covers K=64 as two 32-K sub-stages of identical layout.
constexpr int kRowB  = 80;                 // padded LDS row stride
constexpr int kSub   = 192 * kRowB;        // sub-stage: A(128 rows) + B(64 rows) = 15360 B
constexpr int kStage = 2 * kSub;           // K=64 stage = 30720 B
static_assert(kSub == 15360, "sub-stage size");

// ---------------- CDNA5 async-to-LDS primitives (inline asm; ASYNCcnt) ----------------

DEV void async_cp_b128(unsigned lds_off, const void* gptr) {
  // per-lane LDS dest (VDST), per-lane 64-bit global address (VADDR), GV mode
  asm volatile("global_load_async_to_lds_b128 %0, %1, off"
               :: "v"(lds_off), "v"(gptr) : "memory");
}
DEV void wait_async_le6() { asm volatile("s_wait_asynccnt 0x6" ::: "memory"); }
DEV void wait_async_0()   { asm volatile("s_wait_asynccnt 0x0" ::: "memory"); }

// ---------------- helpers ----------------

DEV bf16 f2bf(float f) {
  union { float f; unsigned u; } in; in.f = f;
  unsigned r = in.u + 0x7FFFu + ((in.u >> 16) & 1u);  // round-to-nearest-even
  union { unsigned short s; bf16 h; } out;
  out.s = (unsigned short)(r >> 16);
  return out.h;
}

// Read one 16x32 bf16 fragment from a padded LDS tile (row stride kRowB bytes).
// CDNA5 16-bit A/B operand layout:
//   lanes 0..15  : row = lane,    K groups {0..7}  and {16..23}
//   lanes 16..31 : row = lane-16, K groups {8..15} and {24..31}
DEV v16bf lds_frag(const unsigned char* p) {
  const int lane = (int)(threadIdx.x & 31);
  const unsigned char* q = p + (lane & 15) * kRowB + ((lane >> 4) << 4);
  v8bf lo = *(const v8bf*)(q);
  v8bf hi = *(const v8bf*)(q + 32);
  return __builtin_shufflevector(lo, hi, 0,1,2,3,4,5,6,7,8,9,10,11,12,13,14,15);
}

DEV v8f wmma_bf16f32(v16bf a, v16bf b, v8f c) {
  return __builtin_amdgcn_wmma_f32_16x16x32_bf16(
      false, a, false, b, (short)0, c, false, false);
}

// Stage a row-major bf16 tile (rows x 32K) into padded LDS via async copies.
// chunks = rows*4 (16B each); all 256 threads participate uniformly.
DEV void stage_tile(const bf16* __restrict__ g, int ld, int row0, int k0,
                    unsigned lds_base, int chunks, int tid) {
  for (int i = tid; i < chunks; i += 256) {
    int r = i >> 2, c = i & 3;
    const bf16* gp = g + (size_t)(row0 + r) * (size_t)ld + (size_t)(k0 + c * 8);
    async_cp_b128(lds_base + (unsigned)(r * kRowB + c * 16), gp);
  }
}

// ---------------- fp32 -> bf16 conversion ----------------

__global__ __launch_bounds__(256) void cvt_f32_bf16(
    const float* __restrict__ in, bf16* __restrict__ out, int n) {
  int i = (int)(blockIdx.x * blockDim.x + threadIdx.x);
  if (i < n) out[i] = f2bf(in[i]);
}

// ---------------- gate: softmax + top-2 -> dense combine weights ----------------

__global__ __launch_bounds__(256) void moe_gate(
    const float* __restrict__ x,       // [T, D] fp32
    const float* __restrict__ gw,      // [E, D]
    const float* __restrict__ gb,      // [E]
    float* __restrict__ gd) {          // [T, E] dense gate weights out
  __shared__ float gws[kE * kDim];     // 32 KB
  __shared__ float xs[256][17];        // padded 16-wide K chunk
  const int tid = (int)threadIdx.x;
  const int t0  = (int)blockIdx.x * 256;

  for (int i = tid; i < kE * kDim; i += 256) gws[i] = gw[i];
  __syncthreads();

  float acc[kE];
#pragma unroll
  for (int e = 0; e < kE; ++e) acc[e] = gb[e];

  for (int k0 = 0; k0 < kDim; k0 += 16) {
    __syncthreads();
    for (int i = tid; i < 256 * 16; i += 256) {
      int tt = i >> 4, kk = i & 15;
      xs[tt][kk] = x[(size_t)(t0 + tt) * kDim + k0 + kk];
    }
    __syncthreads();
#pragma unroll
    for (int kk = 0; kk < 16; ++kk) {
      float xv = xs[tid][kk];
#pragma unroll
      for (int e = 0; e < kE; ++e) acc[e] = fmaf(xv, gws[e * kDim + k0 + kk], acc[e]);
    }
  }

  float m = acc[0];
#pragma unroll
  for (int e = 1; e < kE; ++e) m = fmaxf(m, acc[e]);
  float s = 0.f, p[kE];
#pragma unroll
  for (int e = 0; e < kE; ++e) { p[e] = __expf(acc[e] - m); s += p[e]; }
  float inv = 1.f / s;
#pragma unroll
  for (int e = 0; e < kE; ++e) p[e] *= inv;

  int i1 = 0; float v1 = p[0];
#pragma unroll
  for (int e = 1; e < kE; ++e) if (p[e] > v1) { v1 = p[e]; i1 = e; }
  int i2 = -1; float v2 = -1.f;
#pragma unroll
  for (int e = 0; e < kE; ++e) if (e != i1 && p[e] > v2) { v2 = p[e]; i2 = e; }

#pragma unroll
  for (int e = 0; e < kE; ++e) {
    float wv = (e == i1) ? v1 : (e == i2) ? v2 : 0.f;  // ROUTE_SCALE = 1
    gd[(size_t)(t0 + tid) * kE + e] = wv;
  }
}

// ---------------- fused up-proj dual GEMM + SwiGLU (async double-buffered, K=64 stages) ----
// h[t, n] = silu(x @ w1^T + b1) * (x @ w3^T + b3)
// block = 256 threads (8 waves); block tile 128(M) x 32(N); wave = 16 x 32, both matrices.

__global__ __launch_bounds__(256) void moe_gemm_h(
    const bf16* __restrict__ xb,   // [T, D] bf16
    const bf16* __restrict__ w1,   // [inter, D] bf16
    const bf16* __restrict__ w3,   // [inter, D] bf16
    const float* __restrict__ b1,  // [inter]
    const float* __restrict__ b3,  // [inter]
    bf16* __restrict__ h,          // [T, inter] bf16
    int inter) {
  __shared__ __attribute__((aligned(16))) unsigned char smem[2 * kStage];  // 60 KB
  const int tid  = (int)threadIdx.x;
  const int lane = tid & 31;
  const int wid  = tid >> 5;
  const int trow0 = (int)blockIdx.y * 128;
  const int col0  = (int)blockIdx.x * 32;
  const unsigned base = (unsigned)(size_t)smem;

  // per K=64 stage: each thread issues exactly 6 async b128 copies
  // (A: 2 per sub-stage, B1 or B3: 1 per sub-stage)
  auto issue = [&](int k0, unsigned sb) {
#pragma unroll
    for (int kg = 0; kg < 2; ++kg) {
      unsigned sg = sb + (unsigned)(kg * kSub);
      int kk = k0 + kg * 32;
      stage_tile(xb, kDim, trow0, kk, sg, 512, tid);
      if (tid < 128) {
        int r = tid >> 2, c = tid & 3;
        async_cp_b128(sg + (unsigned)(128 * kRowB + r * kRowB + c * 16),
                      w1 + (size_t)(col0 + r) * kDim + (kk + c * 8));
      } else {
        int i = tid - 128, r = i >> 2, c = i & 3;
        async_cp_b128(sg + (unsigned)(160 * kRowB + r * kRowB + c * 16),
                      w3 + (size_t)(col0 + r) * kDim + (kk + c * 8));
      }
    }
  };

  v8f a1[2] = {};
  v8f a3[2] = {};

  const int nStages = kDim / 64;                  // 16
  issue(0, base);
  for (int s = 0; s < nStages; ++s) {
    if (s + 1 < nStages) {
      issue((s + 1) * 64, base + (unsigned)(((s + 1) & 1) * kStage));
      wait_async_le6();                           // current stage complete
    } else {
      wait_async_0();
    }
    __syncthreads();

    const unsigned char* sm = smem + (s & 1) * kStage;
#pragma unroll
    for (int kg = 0; kg < 2; ++kg) {
      const unsigned char* sg = sm + kg * kSub;
      v16bf A   = lds_frag(sg + wid * 16 * kRowB);
      v16bf B10 = lds_frag(sg + 128 * kRowB);
      v16bf B11 = lds_frag(sg + 144 * kRowB);
      v16bf B30 = lds_frag(sg + 160 * kRowB);
      v16bf B31 = lds_frag(sg + 176 * kRowB);
      a1[0] = wmma_bf16f32(A, B10, a1[0]);
      a1[1] = wmma_bf16f32(A, B11, a1[1]);
      a3[0] = wmma_bf16f32(A, B30, a3[0]);
      a3[1] = wmma_bf16f32(A, B31, a3[1]);
    }
    __syncthreads();                              // done reading this buffer
  }

  const int half = lane >> 4;
  const int c    = lane & 15;
  const int row0 = trow0 + wid * 16;
#pragma unroll
  for (int sIdx = 0; sIdx < 2; ++sIdx) {
    int col = col0 + sIdx * 16 + c;
    float bb1 = b1[col], bb3 = b3[col];
#pragma unroll
    for (int j = 0; j < 8; ++j) {
      int row = row0 + half * 8 + j;
      float v1 = a1[sIdx][j] + bb1;
      float v3 = a3[sIdx][j] + bb3;
      float sig = 1.f / (1.f + __expf(-v1));      // silu(v1) = v1*sigmoid(v1)
      h[(size_t)row * inter + col] = f2bf(v1 * sig * v3);
    }
  }
}

// ---------------- down-proj GEMM + gated combine (async double-buffered, K=64 stages) -----
// out[t, n] (+)= scale[t] * (h @ w2^T + b2)
// block = 256 threads (8 waves, 4Mx2N); block tile 128(M) x 64(N); wave = 32 x 32.

__global__ __launch_bounds__(256) void moe_gemm_out(
    const bf16* __restrict__ hbuf, // [T, K] bf16
    const bf16* __restrict__ w2,   // [D, K] bf16
    const float* __restrict__ b2,  // [D]
    const float* __restrict__ gate,// per-token scale (stride kE), or nullptr -> 1.0
    float* __restrict__ out,       // [T, D] fp32
    int K, int accumulate) {
  __shared__ __attribute__((aligned(16))) unsigned char smem[2 * kStage];  // 60 KB
  const int tid  = (int)threadIdx.x;
  const int lane = tid & 31;
  const int wid  = tid >> 5;
  const int mw   = wid & 3;                       // wave M index (0..3)
  const int nw   = wid >> 2;                      // wave N index (0..1)
  const int trow0 = (int)blockIdx.y * 128;
  const int col0  = (int)blockIdx.x * 64;
  const unsigned base = (unsigned)(size_t)smem;

  // per K=64 stage: each thread issues exactly 6 async b128 copies (A: 4, B: 2)
  auto issue = [&](int k0, unsigned sb) {
#pragma unroll
    for (int kg = 0; kg < 2; ++kg) {
      unsigned sg = sb + (unsigned)(kg * kSub);
      int kk = k0 + kg * 32;
      stage_tile(hbuf, K, trow0, kk, sg, 512, tid);
      int r = tid >> 2, c = tid & 3;
      async_cp_b128(sg + (unsigned)(128 * kRowB + r * kRowB + c * 16),
                    w2 + (size_t)(col0 + r) * (size_t)K + (kk + c * 8));
    }
  };

  v8f acc[2][2] = {};

  const int nStages = K / 64;                     // 16 or 32
  issue(0, base);
  for (int s = 0; s < nStages; ++s) {
    if (s + 1 < nStages) {
      issue((s + 1) * 64, base + (unsigned)(((s + 1) & 1) * kStage));
      wait_async_le6();
    } else {
      wait_async_0();
    }
    __syncthreads();

    const unsigned char* sm = smem + (s & 1) * kStage;
#pragma unroll
    for (int kg = 0; kg < 2; ++kg) {
      const unsigned char* sg = sm + kg * kSub;
      v16bf A0 = lds_frag(sg + (mw * 32 + 0)  * kRowB);
      v16bf A1 = lds_frag(sg + (mw * 32 + 16) * kRowB);
      v16bf B0 = lds_frag(sg + (128 + nw * 32 + 0)  * kRowB);
      v16bf B1 = lds_frag(sg + (128 + nw * 32 + 16) * kRowB);
      acc[0][0] = wmma_bf16f32(A0, B0, acc[0][0]);
      acc[0][1] = wmma_bf16f32(A0, B1, acc[0][1]);
      acc[1][0] = wmma_bf16f32(A1, B0, acc[1][0]);
      acc[1][1] = wmma_bf16f32(A1, B1, acc[1][1]);
    }
    __syncthreads();
  }

  const int half = lane >> 4;
  const int c    = lane & 15;
#pragma unroll
  for (int mi = 0; mi < 2; ++mi) {
#pragma unroll
    for (int j = 0; j < 8; ++j) {
      int row = trow0 + mw * 32 + mi * 16 + half * 8 + j;
      float scale = gate ? gate[(size_t)row * kE] : 1.0f;
#pragma unroll
      for (int ni = 0; ni < 2; ++ni) {
        int col = col0 + nw * 32 + ni * 16 + c;
        float v = scale * (acc[mi][ni][j] + b2[col]);
        size_t o = (size_t)row * kDim + col;
        if (accumulate) out[o] += v; else out[o] = v;
      }
    }
  }
}

// ---------------- host-side orchestration ----------------

extern "C" void kernel_launch(void* const* d_in, const int* in_sizes, int n_in,
                              void* d_out, int out_size, void* d_ws, size_t ws_size,
                              hipStream_t stream) {
  (void)in_sizes; (void)n_in; (void)out_size; (void)ws_size;

  const float* x      = (const float*)d_in[0];   // [T, D]
  const float* gate_w = (const float*)d_in[1];   // [E, D]
  const float* gate_b = (const float*)d_in[2];   // [E]
  const float* w1     = (const float*)d_in[3];   // [E, INTER, D]
  const float* b1     = (const float*)d_in[4];   // [E, INTER]
  const float* w3     = (const float*)d_in[5];   // [E, INTER, D]
  const float* b3     = (const float*)d_in[6];   // [E, INTER]
  const float* w2     = (const float*)d_in[7];   // [E, D, INTER]
  const float* b2     = (const float*)d_in[8];   // [E, D]
  const float* ws1    = (const float*)d_in[9];   // [SI, D]
  const float* bs1    = (const float*)d_in[10];  // [SI]
  const float* ws3    = (const float*)d_in[11];  // [SI, D]
  const float* bs3    = (const float*)d_in[12];  // [SI]
  const float* ws2    = (const float*)d_in[13];  // [D, SI]
  const float* bs2    = (const float*)d_in[14];  // [D]
  float* out          = (float*)d_out;           // [T, D]

  char* wp = (char*)d_ws;
  auto alloc = [&](size_t bytes) -> char* {
    char* p = wp; wp += (bytes + 255) & ~(size_t)255; return p;
  };
  bf16*  xb    = (bf16*) alloc((size_t)kT * kDim * 2);
  bf16*  ws1b  = (bf16*) alloc((size_t)kSharedInter * kDim * 2);
  bf16*  ws3b  = (bf16*) alloc((size_t)kSharedInter * kDim * 2);
  bf16*  ws2b  = (bf16*) alloc((size_t)kDim * kSharedInter * 2);
  bf16*  we1b  = (bf16*) alloc((size_t)kInter * kDim * 2);
  bf16*  we3b  = (bf16*) alloc((size_t)kInter * kDim * 2);
  bf16*  we2b  = (bf16*) alloc((size_t)kDim * kInter * 2);
  float* gd    = (float*)alloc((size_t)kT * kE * 4);
  bf16*  hbuf  = (bf16*) alloc((size_t)kT * kSharedInter * 2);

  auto cvt = [&](const float* src, bf16* dst, size_t n) {
    cvt_f32_bf16<<<(unsigned)((n + 255) / 256), 256, 0, stream>>>(src, dst, (int)n);
  };

  // 1) conversions for x + shared-expert weights
  cvt(x,   xb,   (size_t)kT * kDim);
  cvt(ws1, ws1b, (size_t)kSharedInter * kDim);
  cvt(ws3, ws3b, (size_t)kSharedInter * kDim);
  cvt(ws2, ws2b, (size_t)kDim * kSharedInter);

  // 2) gate -> dense combine weights [T, E]
  moe_gate<<<kT / 256, 256, 0, stream>>>(x, gate_w, gate_b, gd);

  // 3) shared expert: writes out (initializes it)
  moe_gemm_h<<<dim3(kSharedInter / 32, kT / 128), 256, 0, stream>>>(
      xb, ws1b, ws3b, bs1, bs3, hbuf, kSharedInter);
  moe_gemm_out<<<dim3(kDim / 64, kT / 128), 256, 0, stream>>>(
      hbuf, ws2b, bs2, nullptr, out, kSharedInter, /*accumulate=*/0);

  // 4) routed experts: dense compute, gate-weighted accumulate into out
  for (int e = 0; e < kE; ++e) {
    cvt(w1 + (size_t)e * kInter * kDim, we1b, (size_t)kInter * kDim);
    cvt(w3 + (size_t)e * kInter * kDim, we3b, (size_t)kInter * kDim);
    cvt(w2 + (size_t)e * kDim * kInter, we2b, (size_t)kDim * kInter);
    moe_gemm_h<<<dim3(kInter / 32, kT / 128), 256, 0, stream>>>(
        xb, we1b, we3b, b1 + (size_t)e * kInter, b3 + (size_t)e * kInter, hbuf, kInter);
    moe_gemm_out<<<dim3(kDim / 64, kT / 128), 256, 0, stream>>>(
        hbuf, we2b, b2 + (size_t)e * kDim, gd + e, out, kInter, /*accumulate=*/1);
  }
}
